// GSH_44066364457373
// MI455X (gfx1250) — matrix operand order repaired and checked by reading, not compiled, not run on hardware
//
#include <hip/hip_runtime.h>

// ---- CDNA5 WMMA vector types ----
typedef __attribute__((ext_vector_type(16))) _Float16 v16h;
typedef __attribute__((ext_vector_type(8)))  float    v8f;

// Problem constants (match reference)
constexpr int Bb = 2, Lq = 2048, Sk = 2048, Hh = 16, Ee = 64, Dd = 64;
constexpr int QT = 16;                  // queries per workgroup (WMMA M)
constexpr int SC_STRIDE = Sk + 4;       // padded LDS row stride (floats); 2052 % 64 = 4 keeps
                                        // hi/lo half-wave store banks disjoint
constexpr float SCALE = 0.125f;         // 1/sqrt(E) = 2^-3, exact under f16 conversion

__global__ __launch_bounds__(128)
void sparsemax_attn_f16wmma(const float* __restrict__ Q,
                            const float* __restrict__ K,
                            const float* __restrict__ V,
                            float* __restrict__ O)
{
    extern __shared__ float sc[];       // [QT][SC_STRIDE] score strip, ~128 KB
    const int tid  = threadIdx.x;
    const int wave = tid >> 5;          // 0..3
    const int lane = tid & 31;
    const bool hi  = lane >= 16;
    const int l16  = lane & 15;

    const int wg    = blockIdx.x;
    const int qTile = wg % (Lq / QT);
    const int h     = (wg / (Lq / QT)) % Hh;
    const int b     =  wg / ((Lq / QT) * Hh);
    const int q0    = qTile * QT;

    // ================= Phase 1: scores = (SCALE*Q) K^T  -> LDS =================
    // A operand in this code's k order: lane<16 row=lane holds K eb..eb+15 contiguous,
    // eb = e0 (+16 for hi lanes). SCALE folded into Q here: 2^-3 scaling is exact,
    // so the hot score loop needs no post-multiply.
    v16h aQ[2];
    {
        const int m = l16;
        const float* qrow = Q + (((size_t)b * Lq + q0 + m) * Hh + h) * Ee;
        #pragma unroll
        for (int c = 0; c < 2; ++c) {
            const int eb = 32 * c + (hi ? 16 : 0);
            #pragma unroll
            for (int j = 0; j < 8; ++j) {
                aQ[c][j]     = (_Float16)(qrow[eb + j] * SCALE);
                aQ[c][8 + j] = (_Float16)(qrow[eb + 8 + j] * SCALE);
            }
        }
    }

    // Each wave covers 16-key tiles strided by 4 (32 tiles per wave).
    for (int t = wave; t < Sk / 16; t += 4) {
        const int s0 = t * 16;
        // B operand in the same k order as A: lane<16 -> N=lane, K=eb..eb+15 contiguous (eb=e0 or e0+16)
        const float* krow = K + (((size_t)b * Sk + s0 + l16) * Hh + h) * Ee;
        v8f acc = {};
        #pragma unroll
        for (int c = 0; c < 2; ++c) {
            const int eb = 32 * c + (hi ? 16 : 0);
            v16h bK;
            #pragma unroll
            for (int j = 0; j < 16; ++j) bK[j] = (_Float16)krow[eb + j];
            acc = __builtin_amdgcn_wmma_f32_16x16x32_f16(false, aQ[c], false, bK,
                                                         (short)0, acc, false, false);
        }
        // f32 C/D layout: VGPR r -> (M=r, N=lane) lanes 0-15, (M=8+r, N=lane-16) lanes 16-31
        #pragma unroll
        for (int r = 0; r < 8; ++r) {
            const int mr = r + (hi ? 8 : 0);
            sc[mr * SC_STRIDE + s0 + l16] = acc[r];
        }
    }
    __syncthreads();

    // ================= Phase 2: sparsemax per row (bisection + exact refine) =================
    // 8 lanes per row; reductions via shfl_xor within aligned groups of 8 (same wave, no barriers).
    {
        const int row = tid >> 3;       // 0..15
        const int w   = tid & 7;
        float* z = sc + row * SC_STRIDE;

        float mx = -3.4e38f;
        for (int i = w; i < Sk; i += 8) mx = fmaxf(mx, z[i]);
        mx = fmaxf(mx, __shfl_xor(mx, 1, 32));
        mx = fmaxf(mx, __shfl_xor(mx, 2, 32));
        mx = fmaxf(mx, __shfl_xor(mx, 4, 32));

        // f(tau) = sum(max(z - tau,0)) - 1 is monotone decreasing; root in [mx-1, mx].
        // 26 halvings of a width-1 interval -> ~1.5e-8 < f32 ulp at tau ~ O(1).
        float lo = mx - 1.0f, hg = mx;
        for (int it = 0; it < 26; ++it) {
            const float mid = 0.5f * (lo + hg);
            float ssum = 0.0f;
            for (int i = w; i < Sk; i += 8) ssum += fmaxf(z[i] - mid, 0.0f);
            ssum += __shfl_xor(ssum, 1, 32);
            ssum += __shfl_xor(ssum, 2, 32);
            ssum += __shfl_xor(ssum, 4, 32);
            if (ssum >= 1.0f) lo = mid; else hg = mid;
        }
        // Exact tau from the resolved support set: tau = (sum_{z>taub} z - 1)/k
        const float taub = 0.5f * (lo + hg);
        float ssum = 0.0f, cnt = 0.0f;
        for (int i = w; i < Sk; i += 8) {
            const float zi = z[i];
            if (zi > taub) { ssum += zi; cnt += 1.0f; }
        }
        ssum += __shfl_xor(ssum, 1, 32);  cnt += __shfl_xor(cnt, 1, 32);
        ssum += __shfl_xor(ssum, 2, 32);  cnt += __shfl_xor(cnt, 2, 32);
        ssum += __shfl_xor(ssum, 4, 32);  cnt += __shfl_xor(cnt, 4, 32);
        const float tau = (cnt > 0.0f) ? (ssum - 1.0f) / cnt : taub;
        for (int i = w; i < Sk; i += 8) z[i] = fmaxf(z[i] - tau, 0.0f);
    }
    __syncthreads();

    // ================= Phase 3: out[16x64] = A[16xS] @ V[SxD], wave owns d0 = 16*wave ======
    {
        const int d0 = wave * 16;
        v8f acc = {};
        for (int s0 = 0; s0 < Sk; s0 += 32) {
            // A fragment from LDS probabilities (f32 -> f16); one 64B-contiguous run per lane
            const float* prow = sc + l16 * SC_STRIDE;
            const int sb = s0 + (hi ? 16 : 0);
            v16h aA;
            #pragma unroll
            for (int j = 0; j < 8; ++j) {
                aA[j]     = (_Float16)prow[sb + j];
                aA[8 + j] = (_Float16)prow[sb + 8 + j];
            }
            // B fragment from V: lane<16 -> N=d0+lane, K=s0..s0+15; hi lanes K=s0+16..s0+31.
            // Each load instruction covers two 64B-contiguous cache lines across the half-waves.
            const int sbv = s0 + (hi ? 16 : 0);
            const float* vcol = V + (((size_t)b * Sk + sbv) * Hh + h) * Dd + d0 + l16;
            v16h bV;
            #pragma unroll
            for (int j = 0; j < 16; ++j) bV[j] = (_Float16)vcol[(size_t)j * Hh * Dd];
            acc = __builtin_amdgcn_wmma_f32_16x16x32_f16(false, aA, false, bV,
                                                         (short)0, acc, false, false);
        }
        #pragma unroll
        for (int r = 0; r < 8; ++r) {
            const int mr = r + (hi ? 8 : 0);
            O[(((size_t)b * Lq + q0 + mr) * Hh + h) * Dd + d0 + l16] = acc[r];
        }
    }
}

extern "C" void kernel_launch(void* const* d_in, const int* in_sizes, int n_in,
                              void* d_out, int out_size, void* d_ws, size_t ws_size,
                              hipStream_t stream)
{
    (void)in_sizes; (void)n_in; (void)out_size; (void)d_ws; (void)ws_size;
    const float* Q = (const float*)d_in[0];
    const float* K = (const float*)d_in[1];
    const float* V = (const float*)d_in[2];
    float* O = (float*)d_out;

    const dim3 grid(Bb * Hh * (Lq / QT));   // 4096 workgroups
    const dim3 block(128);                  // 4 wave32
    const size_t shmem = (size_t)QT * SC_STRIDE * sizeof(float);  // ~128.25 KB (CDNA5: 320 KB/WGP)
    sparsemax_attn_f16wmma<<<grid, block, shmem, stream>>>(Q, K, V, O);
}
